// WeightedCosine_29703993819992
// MI455X (gfx1250) — compile-verified, matrix operand order checked by reading, not run on hardware
//
#include <hip/hip_runtime.h>

typedef __attribute__((ext_vector_type(16))) _Float16 v16h;
typedef __attribute__((ext_vector_type(8)))  _Float16 v8h;
typedef __attribute__((ext_vector_type(4)))  _Float16 v4h;
typedef __attribute__((ext_vector_type(8)))  float    v8f;
typedef __attribute__((ext_vector_type(4)))  float    v4f;

#define N_ROWS  8192
#define D_DIM   128
#define P_HEADS 4
#define K_DIM   512   // P_HEADS * D_DIM

// ---------------------------------------------------------------------------
// Stage 1: build A[n, p*128+d] = 0.5 * (emb[n,d]*w[p,d]) / max(||emb[n]*w[p]||, eps)
// in f16.  The 0.5 on both GEMM operands folds the mean over 4 heads.
// One thread per (n,p) pair; all data is tiny and L2-resident.
// ---------------------------------------------------------------------------
__global__ __launch_bounds__(256) void wc_normalize(const float* __restrict__ emb,
                                                    const float* __restrict__ wgt,
                                                    _Float16* __restrict__ A) {
    const int idx = blockIdx.x * blockDim.x + threadIdx.x;   // 0 .. N*P-1
    const int n = idx >> 2;
    const int p = idx & 3;
    const float* __restrict__ e = emb + (size_t)n * D_DIM;
    const float* __restrict__ w = wgt + (size_t)p * D_DIM;

    float ss = 0.0f;
#pragma unroll
    for (int d = 0; d < D_DIM; d += 4) {
        v4f ev = *(const v4f*)(e + d);
        v4f wv = *(const v4f*)(w + d);
        v4f pv = ev * wv;
        ss += pv.x * pv.x + pv.y * pv.y + pv.z * pv.z + pv.w * pv.w;
    }
    // torch F.normalize: x / max(||x||, eps); fold 0.5 for the head-mean.
    const float inv = 0.5f / fmaxf(sqrtf(ss), 1e-12f);

    _Float16* __restrict__ out = A + (size_t)n * K_DIM + p * D_DIM;
#pragma unroll
    for (int d = 0; d < D_DIM; d += 4) {
        v4f ev = *(const v4f*)(e + d);
        v4f wv = *(const v4f*)(w + d);
        v4f pv = ev * wv;
        v4h hv;
        hv.x = (_Float16)(pv.x * inv);
        hv.y = (_Float16)(pv.y * inv);
        hv.z = (_Float16)(pv.z * inv);
        hv.w = (_Float16)(pv.w * inv);
        *(v4h*)(out + d) = hv;   // 8B aligned (row offsets are multiples of 256B)
    }
}

// ---------------------------------------------------------------------------
// Fragment loader for V_WMMA_F32_16X16X32_F16 (row-major f16 source, ld = K_DIM).
// ISA 16-bit A layout (identical addressing works for the B operand of A@A^T):
//   lane L<16 : row = base+L,    halves = K[kc+0..7]  then K[kc+16..23]
//   lane L>=16: row = base+L-16, halves = K[kc+8..15] then K[kc+24..31]
// => two 16-byte loads per lane per fragment.
// ---------------------------------------------------------------------------
__device__ __forceinline__ v16h load_frag(const _Float16* __restrict__ A,
                                          int row, int kc, int khalf) {
    const _Float16* p = A + (size_t)row * K_DIM + kc + khalf;
    v8h lo = *(const v8h*)(p);
    v8h hi = *(const v8h*)(p + 16);
    return __builtin_shufflevector(lo, hi, 0, 1, 2, 3, 4, 5, 6, 7,
                                           8, 9, 10, 11, 12, 13, 14, 15);
}

// ---------------------------------------------------------------------------
// Stage 2: C = A @ A^T, M = N = 8192, K = 512, fp32 accumulate/output.
// 256 threads = 8 waves; WG tile 128(M) x 256(N); waves 2(M) x 4(N); each wave
// owns a 64x64 sub-tile = 4x4 WMMA tiles -> 16 v_wmma_f32_16x16x32_f16 per
// K-step of 32 against 8 fragment loads (~32 FLOP/B of cache traffic).
// Registers: 128 acc + 64 fragment VGPRs, ~210 total -> 2 waves/SIMD.
// ---------------------------------------------------------------------------
__global__ __launch_bounds__(256) void wc_gemm(const _Float16* __restrict__ A,
                                               float* __restrict__ C) {
    const int lane  = threadIdx.x & 31;
    const int wave  = threadIdx.x >> 5;   // 0..7
    const int wm    = wave >> 2;          // 2 M-groups
    const int wn    = wave & 3;           // 4 N-groups
    const int mbase = blockIdx.x * 128 + wm * 64;
    const int nbase = blockIdx.y * 256 + wn * 64;

    const int lrow  = lane & 15;
    const int khalf = (lane >> 4) * 8;    // 0 or 8

    v8f acc[4][4];
#pragma unroll
    for (int tm = 0; tm < 4; ++tm)
#pragma unroll
        for (int tn = 0; tn < 4; ++tn)
            acc[tm][tn] = (v8f){0.f, 0.f, 0.f, 0.f, 0.f, 0.f, 0.f, 0.f};

    for (int kc = 0; kc < K_DIM; kc += 32) {
        v16h a[4], b[4];
#pragma unroll
        for (int tm = 0; tm < 4; ++tm) {
            a[tm] = load_frag(A, mbase + tm * 16 + lrow, kc, khalf);
            // hint the next K chunk of this row into cache (global_prefetch_b8)
            __builtin_prefetch(A + (size_t)(mbase + tm * 16 + lrow) * K_DIM + kc + 32, 0, 1);
        }
#pragma unroll
        for (int tn = 0; tn < 4; ++tn)
            b[tn] = load_frag(A, nbase + tn * 16 + lrow, kc, khalf);

#pragma unroll
        for (int tm = 0; tm < 4; ++tm)
#pragma unroll
            for (int tn = 0; tn < 4; ++tn)
                acc[tm][tn] = __builtin_amdgcn_wmma_f32_16x16x32_f16(
                    /*neg_a=*/false, a[tm], /*neg_b=*/false, b[tn],
                    /*c_mod=*/(short)0, acc[tm][tn],
                    /*reuse_a=*/false, /*reuse_b=*/false);
    }

    // C/D layout: VGPR r -> row (tile + r + 8*(lane>=16)), col = tile + (lane&15)
#pragma unroll
    for (int tm = 0; tm < 4; ++tm) {
        const int rbase = mbase + tm * 16 + khalf;
#pragma unroll
        for (int tn = 0; tn < 4; ++tn) {
            const int col = nbase + tn * 16 + lrow;
#pragma unroll
            for (int r = 0; r < 8; ++r)
                C[(size_t)(rbase + r) * N_ROWS + col] = acc[tm][tn][r];
        }
    }
}

extern "C" void kernel_launch(void* const* d_in, const int* in_sizes, int n_in,
                              void* d_out, int out_size, void* d_ws, size_t ws_size,
                              hipStream_t stream) {
    (void)in_sizes; (void)n_in; (void)out_size; (void)ws_size;
    const float* emb = (const float*)d_in[0];   // (8192, 128) fp32
    const float* wgt = (const float*)d_in[1];   // (4, 128) fp32
    float* out = (float*)d_out;                 // (8192, 8192) fp32
    _Float16* A = (_Float16*)d_ws;              // (8192, 512) f16, 8 MB scratch

    wc_normalize<<<(N_ROWS * P_HEADS) / 256, 256, 0, stream>>>(emb, wgt, A);

    dim3 grid(N_ROWS / 128, N_ROWS / 256);      // 64 x 32 workgroups
    wc_gemm<<<grid, 256, 0, stream>>>(A, out);
}
